// Decoder_7928509628749
// MI455X (gfx1250) — matrix operand order repaired and checked by reading, not compiled, not run on hardware
//
#include <hip/hip_runtime.h>
#include <hip/hip_bf16.h>

// Problem constants
#define BB   8
#define SS   256
#define UU   8
#define TT   10
#define VV   30522
#define HH   768
#define RR   64          // B*U batched rows
#define GG   2304        // 3*H
#define NTIL 1908        // ceil(V/16)
#define VPAD (NTIL*16)   // 30528

typedef __attribute__((ext_vector_type(16))) __bf16 v16bf;
typedef __attribute__((ext_vector_type(8)))  __bf16 v8bf;
typedef __attribute__((ext_vector_type(8)))  float  v8f;

__device__ __forceinline__ unsigned short f2bf(float f) {
    union { float f; unsigned u; } v; v.f = f;
    unsigned u = v.u;
    unsigned r = u + 0x7FFFu + ((u >> 16) & 1u);   // round-to-nearest-even
    return (unsigned short)(r >> 16);
}

// ---- emb fp32 -> bf16, padded to VPAD rows (pad rows = 0) ----------------
__global__ void k_embcvt(const float* __restrict__ emb, unsigned short* __restrict__ ebf) {
    size_t idx = (size_t)blockIdx.x * blockDim.x + threadIdx.x;
    if (idx >= (size_t)VPAD * HH) return;
    size_t row = idx / HH;
    ebf[idx] = (row < VV) ? f2bf(emb[idx]) : (unsigned short)0;
}

// ---- init x = update_vecs, h = broadcast(enc_hidden) ---------------------
__global__ void k_init(const float* __restrict__ upd, const float* __restrict__ eh,
                       float* __restrict__ x, float* __restrict__ h) {
    int idx = blockIdx.x * blockDim.x + threadIdx.x;
    if (idx >= RR * HH) return;
    int r = idx / HH, k = idx - r * HH;
    x[idx] = upd[idx];               // update_vecs is [B,U,H] flat == r*H+k
    h[idx] = eh[(r / UU) * HH + k];  // enc_hidden [B,H]
}

// ---- GRU gate pre-activations gi = x@W_ih^T + b_ih ; gh = h@W_hh^T + b_hh
__global__ void k_gru_gates(const float* __restrict__ x, const float* __restrict__ h,
                            const float* __restrict__ Wih, const float* __restrict__ Whh,
                            const float* __restrict__ bih, const float* __restrict__ bhh,
                            float* __restrict__ gi, float* __restrict__ gh) {
    int idx = blockIdx.x * blockDim.x + threadIdx.x;
    if (idx >= RR * GG) return;
    int r = idx / GG, j = idx - r * GG;
    const float* xr = x + (size_t)r * HH;
    const float* hr = h + (size_t)r * HH;
    const float* wi = Wih + (size_t)j * HH;
    const float* wh = Whh + (size_t)j * HH;
    float ai = 0.f, ah = 0.f;
    for (int k = 0; k < HH; k += 4) {
        float4 xv = *(const float4*)(xr + k);
        float4 hv = *(const float4*)(hr + k);
        float4 wv = *(const float4*)(wi + k);
        float4 uv = *(const float4*)(wh + k);
        ai += xv.x*wv.x + xv.y*wv.y + xv.z*wv.z + xv.w*wv.w;
        ah += hv.x*uv.x + hv.y*uv.y + hv.z*uv.z + hv.w*uv.w;
    }
    gi[idx] = ai + bih[j];
    gh[idx] = ah + bhh[j];
}

// ---- GRU combine -> new h (f32 + bf16), also writes hs output slice ------
__global__ void k_gru_combine(const float* __restrict__ gi, const float* __restrict__ gh,
                              float* __restrict__ h, unsigned short* __restrict__ hbf,
                              float* __restrict__ hsBase, int t) {
    int idx = blockIdx.x * blockDim.x + threadIdx.x;
    if (idx >= RR * HH) return;
    int r = idx / HH, k = idx - r * HH;
    size_t base = (size_t)r * GG;
    float i_r = gi[base + k], i_z = gi[base + HH + k], i_n = gi[base + 2*HH + k];
    float h_r = gh[base + k], h_z = gh[base + HH + k], h_n = gh[base + 2*HH + k];
    float rg = 1.f / (1.f + expf(-(i_r + h_r)));
    float zg = 1.f / (1.f + expf(-(i_z + h_z)));
    float ng = tanhf(i_n + rg * h_n);
    float hn = (1.f - zg) * ng + zg * h[idx];
    h[idx]   = hn;
    hbf[idx] = f2bf(hn);
    hsBase[(size_t)(r * TT + t) * HH + k] = hn;
}

// ---- attention: scores->softmax->prob, ctx = prob @ enc_seq --------------
__global__ void k_attn(const float* __restrict__ h, const float* __restrict__ enc,
                       const int* __restrict__ mask,
                       float* __restrict__ prob, float* __restrict__ ctx) {
    int r = blockIdx.x;            // 0..63
    int s = threadIdx.x;           // 0..255
    int b = r >> 3;                // r / U
    __shared__ float sm[SS];
    __shared__ float sp[SS];
    const float* hr = h + (size_t)r * HH;
    const float* es = enc + ((size_t)b * SS + s) * HH;
    float acc = 0.f;
    for (int k = 0; k < HH; k += 4) {
        float4 hv = *(const float4*)(hr + k);
        float4 ev = *(const float4*)(es + k);
        acc += hv.x*ev.x + hv.y*ev.y + hv.z*ev.z + hv.w*ev.w;
    }
    acc += ((float)mask[b * SS + s] - 1.0f) * 4294967296.0f;
    sm[s] = acc; __syncthreads();
    for (int off = 128; off > 0; off >>= 1) { if (s < off) sm[s] = fmaxf(sm[s], sm[s+off]); __syncthreads(); }
    float mx = sm[0]; __syncthreads();
    float e = expf(acc - mx);
    sm[s] = e; __syncthreads();
    for (int off = 128; off > 0; off >>= 1) { if (s < off) sm[s] += sm[s+off]; __syncthreads(); }
    float p = e / sm[0];
    sp[s] = p;
    prob[r * SS + s] = p;
    __syncthreads();
    const float* eb = enc + (size_t)b * SS * HH;
    for (int hh = s; hh < HH; hh += 256) {
        float c = 0.f;
        for (int ss2 = 0; ss2 < SS; ++ss2) c += sp[ss2] * eb[(size_t)ss2 * HH + hh];
        ctx[(size_t)r * HH + hh] = c;
    }
}

// ---- gate = sigmoid([h,ctx,x] . gate_w + gate_b) -------------------------
__global__ void k_gate(const float* __restrict__ h, const float* __restrict__ ctx,
                       const float* __restrict__ x, const float* __restrict__ gw,
                       const float* __restrict__ gb, float* __restrict__ gate) {
    int r = blockIdx.x, tid = threadIdx.x;
    __shared__ float sm[256];
    float acc = 0.f;
    for (int j = tid; j < GG; j += 256) {
        float v = (j < HH) ? h[(size_t)r*HH + j]
                : (j < 2*HH) ? ctx[(size_t)r*HH + j - HH]
                             : x[(size_t)r*HH + j - 2*HH];
        acc += v * gw[j];
    }
    sm[tid] = acc; __syncthreads();
    for (int off = 128; off > 0; off >>= 1) { if (tid < off) sm[tid] += sm[tid+off]; __syncthreads(); }
    if (tid == 0) gate[r] = 1.f / (1.f + expf(-(sm[0] + gb[0])));
}

// ---- logits = h_bf16 @ emb_bf16^T via v_wmma_f32_16x16x32_bf16 -----------
// One wave -> one 16x16 C tile. A: 16xK rows of hbf; B: emb rows are B^T.
__global__ void k_gemm_logits(const unsigned short* __restrict__ hbf,
                              const unsigned short* __restrict__ ebf,
                              float* __restrict__ outBase, int t) {
    int wave = (blockIdx.x * blockDim.x + threadIdx.x) >> 5;
    int lane = threadIdx.x & 31;
    int mtile = wave & 3;
    int ntile = wave >> 2;
    if (ntile >= NTIL) return;
    int m0 = mtile << 4;
    int n0 = ntile << 4;
    int lm = lane & 15;
    int hi = lane >> 4;                 // 0 for lanes 0-15, 1 for 16-31
    int ahalf = hi ? 8 : 0;             // A K-half offset
    int bhalf = hi ? 8 : 0;             // B K-half offset
    const unsigned short* arow = hbf + (size_t)(m0 + lm) * HH;
    const unsigned short* brow = ebf + (size_t)(n0 + lm) * HH;
    v8f acc = {};
    for (int kk = 0; kk < HH; kk += 32) {
        const v8bf* pa = (const v8bf*)(arow + kk + ahalf);
        v8bf a0 = pa[0];                        // K = kk+ahalf .. +7
        v8bf a1 = pa[2];                        // K = kk+ahalf+16 .. +23
        const v8bf* pb = (const v8bf*)(brow + kk + bhalf);
        v8bf b0 = pb[0];                        // K = kk+bhalf .. +7
        v8bf b1 = pb[2];                        // K = kk+bhalf+16 .. +23
        if (kk + 192 <= HH) __builtin_prefetch(brow + kk + 160, 0, 0);
        v16bf A = __builtin_shufflevector(a0, a1, 0,1,2,3,4,5,6,7,8,9,10,11,12,13,14,15);
        v16bf Bv = __builtin_shufflevector(b0, b1, 0,1,2,3,4,5,6,7,8,9,10,11,12,13,14,15);
        acc = __builtin_amdgcn_wmma_f32_16x16x32_bf16(
                  false, A, false, Bv, (short)0, acc, false, false);
    }
    int col = n0 + lm;
    if (col < VV) {
        int mbase = m0 + (hi ? 8 : 0);
        #pragma unroll
        for (int v = 0; v < 8; ++v) {
            int r = mbase + v;
            outBase[(size_t)(r * TT + t) * VV + col] = acc[v];
        }
    }
}

// ---- per-row max & sum(exp) over V ---------------------------------------
__global__ void k_rowstats(const float* __restrict__ outBase,
                           float* __restrict__ rmax, float* __restrict__ rsum, int t) {
    int r = blockIdx.x, tid = threadIdx.x;
    const float* row = outBase + (size_t)(r * TT + t) * VV;
    __shared__ float sm[256];
    float mx = -3.4e38f;
    for (int i = tid; i < VV; i += 256) mx = fmaxf(mx, row[i]);
    sm[tid] = mx; __syncthreads();
    for (int off = 128; off > 0; off >>= 1) { if (tid < off) sm[tid] = fmaxf(sm[tid], sm[tid+off]); __syncthreads(); }
    mx = sm[0]; __syncthreads();
    float s = 0.f;
    for (int i = tid; i < VV; i += 256) s += expf(row[i] - mx);
    sm[tid] = s; __syncthreads();
    for (int off = 128; off > 0; off >>= 1) { if (tid < off) sm[tid] += sm[tid+off]; __syncthreads(); }
    if (tid == 0) { rmax[r] = mx; rsum[r] = sm[0]; }
}

// ---- in-place: out = gate * softmax(logits) ------------------------------
__global__ void k_finalize(float* __restrict__ outBase, const float* __restrict__ rmax,
                           const float* __restrict__ rsum, const float* __restrict__ gate, int t) {
    int idx = blockIdx.x * blockDim.x + threadIdx.x;
    if (idx >= RR * VV) return;
    int r = idx / VV, n = idx - r * VV;
    float* p = outBase + (size_t)(r * TT + t) * VV + n;
    *p = gate[r] * expf(*p - rmax[r]) / rsum[r];
}

// ---- pointer scatter: out[ctx_ids] += (1-gate)*prob ----------------------
__global__ void k_scatter(float* __restrict__ outBase, const int* __restrict__ ids,
                          const float* __restrict__ prob, const float* __restrict__ gate, int t) {
    int idx = blockIdx.x * blockDim.x + threadIdx.x;   // 64*256
    if (idx >= RR * SS) return;
    int r = idx >> 8, s = idx & 255;
    int b = r >> 3;
    int id = ids[b * SS + s];
    atomicAdd(outBase + (size_t)(r * TT + t) * VV + id, (1.f - gate[r]) * prob[r * SS + s]);
}

// ---- teacher forcing: x_{t+1} = emb[tgt_ids[:,:,t]] ----------------------
__global__ void k_nextx(const int* __restrict__ tgt, const float* __restrict__ emb,
                        float* __restrict__ x, int t) {
    int idx = blockIdx.x * blockDim.x + threadIdx.x;
    if (idx >= RR * HH) return;
    int r = idx / HH, k = idx - r * HH;
    int tok = tgt[r * TT + t];
    x[idx] = emb[(size_t)tok * HH + k];
}

extern "C" void kernel_launch(void* const* d_in, const int* in_sizes, int n_in,
                              void* d_out, int out_size, void* d_ws, size_t ws_size,
                              hipStream_t stream) {
    (void)in_sizes; (void)n_in; (void)out_size; (void)ws_size;
    const int*   ctx_ids  = (const int*)  d_in[0];
    const int*   ctx_mask = (const int*)  d_in[1];
    const int*   tgt_ids  = (const int*)  d_in[2];
    const float* enc_seq  = (const float*)d_in[3];
    const float* enc_hid  = (const float*)d_in[4];
    const float* upd      = (const float*)d_in[5];
    const float* emb      = (const float*)d_in[6];
    const float* W_ih     = (const float*)d_in[7];
    const float* W_hh     = (const float*)d_in[8];
    const float* b_ih     = (const float*)d_in[9];
    const float* b_hh     = (const float*)d_in[10];
    const float* gate_w   = (const float*)d_in[11];
    const float* gate_b   = (const float*)d_in[12];

    char* ws = (char*)d_ws;
    float*          x    = (float*)(ws + 0);          // 64*768*4   = 196608
    float*          h    = (float*)(ws + 196608);     //             196608
    float*          gi   = (float*)(ws + 393216);     // 64*2304*4 = 589824
    float*          gh   = (float*)(ws + 983040);     //             589824
    float*          ctx  = (float*)(ws + 1572864);    //             196608
    float*          prob = (float*)(ws + 1769472);    // 64*256*4  =  65536
    float*          gate = (float*)(ws + 1835008);    //                256
    float*          rmax = (float*)(ws + 1835264);    //                256
    float*          rsum = (float*)(ws + 1835520);    //                256
    unsigned short* hbf  = (unsigned short*)(ws + 1835776);  // 64*768*2 = 98304
    unsigned short* ebf  = (unsigned short*)(ws + 1934080);  // 30528*768*2

    float* outBase = (float*)d_out;
    float* hsBase  = outBase + (size_t)BB * UU * TT * VV;

    // one-time per launch: emb -> bf16 (padded), init x/h
    k_embcvt<<<(VPAD * HH + 255) / 256, 256, 0, stream>>>(emb, ebf);
    k_init  <<<(RR * HH + 255) / 256, 256, 0, stream>>>(upd, enc_hid, x, h);

    for (int t = 0; t < TT; ++t) {
        k_gru_gates  <<<(RR * GG + 255) / 256, 256, 0, stream>>>(x, h, W_ih, W_hh, b_ih, b_hh, gi, gh);
        k_gru_combine<<<(RR * HH + 255) / 256, 256, 0, stream>>>(gi, gh, h, hbf, hsBase, t);
        k_attn       <<<RR, 256, 0, stream>>>(h, enc_seq, ctx_mask, prob, ctx);
        k_gate       <<<RR, 256, 0, stream>>>(h, ctx, x, gate_w, gate_b, gate);
        k_gemm_logits<<<(4 * NTIL) / 8, 256, 0, stream>>>(hbf, ebf, outBase, t);
        k_rowstats   <<<RR, 256, 0, stream>>>(outBase, rmax, rsum, t);
        k_finalize   <<<(RR * VV + 255) / 256, 256, 0, stream>>>(outBase, rmax, rsum, gate, t);
        k_scatter    <<<(RR * SS + 255) / 256, 256, 0, stream>>>(outBase, ctx_ids, prob, gate, t);
        if (t < TT - 1)
            k_nextx  <<<(RR * HH + 255) / 256, 256, 0, stream>>>(tgt_ids, emb, x, t);
    }
}
